// CrossBlock_18502719111913
// MI455X (gfx1250) — matrix operand order repaired and checked by reading, not compiled, not run on hardware
//
#include <hip/hip_runtime.h>
#include <hip/hip_bf16.h>
#include <math.h>

typedef _Float16 half4  __attribute__((ext_vector_type(4)));
typedef _Float16 half8  __attribute__((ext_vector_type(8)));
typedef _Float16 half16 __attribute__((ext_vector_type(16)));
typedef float    float8 __attribute__((ext_vector_type(8)));
typedef unsigned int uintx4 __attribute__((ext_vector_type(4)));
typedef unsigned int uintx8 __attribute__((ext_vector_type(8)));

#define C_DIM 768
#define F_DIM 3072
#define DH    64
#define NHEAD 12
#define BATCH 2
#define TSEQ  2048
#define MROWS (BATCH*TSEQ)

static __device__ __forceinline__ float8 wmma16(half16 a, half16 b, float8 c) {
  // D = A(16x32 f16) x B(32x16 f16) + C(16x16 f32)
  return __builtin_amdgcn_wmma_f32_16x16x32_f16(false, a, false, b, (short)0, c, false, false);
}

// ---- xor-butterfly all-reduce within 16-lane groups ----
// Prefer v_permlane16_b32 (pure VALU) over __shfl_xor (lowers to ds_bpermute).
#if __has_builtin(__builtin_amdgcn_permlane16)
#define XOR16(OP, v)                                                              \
  {                                                                               \
    const unsigned slo[4] = {0x67452301u, 0x54761032u, 0x32107654u, 0xFEDCBA98u}; \
    const unsigned shi[4] = {0xEFCDAB89u, 0xDCFE98BAu, 0xBA98FEDCu, 0x76543210u}; \
    _Pragma("unroll") for (int s_ = 0; s_ < 4; s_++) {                            \
      unsigned u_ = __builtin_bit_cast(unsigned, v);                              \
      unsigned p_ = __builtin_amdgcn_permlane16(u_, u_, slo[s_], shi[s_], false, false); \
      v = OP(v, __builtin_bit_cast(float, p_));                                   \
    }                                                                             \
  }
#else
#define XOR16(OP, v)                                                              \
  { _Pragma("unroll") for (int d_ = 1; d_ < 16; d_ <<= 1) v = OP(v, __shfl_xor(v, d_, 16)); }
#endif
static __device__ __forceinline__ float fadd_(float a, float b) { return a + b; }

// ---- Tensor Data Mover: DMA a (64 rows x 32 f16) tile from global to LDS ----
// D# group0: count=1, lds_addr, global tile addr (57b), type=2
// D# group1: data_size=1 (2 bytes), tensor_dim0=row_len, tensor_dim1=64,
//            tile_dim0=32, tile_dim1=64, tensor_dim0_stride=row_stride
static __device__ __forceinline__ void tdm_load_tile_64x32(unsigned lds_addr,
                                                           unsigned long long gaddr,
                                                           unsigned row_len_elems,
                                                           unsigned long long row_stride_elems) {
  uintx4 g0;
  g0.x = 1u;                                               // count=1, user mode
  g0.y = lds_addr;                                         // LDS byte address
  g0.z = (unsigned)(gaddr & 0xffffffffu);                  // global addr [31:0]
  g0.w = (unsigned)((gaddr >> 32) & 0x01ffffffu)           // global addr [56:32]
       | (2u << 30);                                       // type = 2 ("image")
  uintx8 g1;
  g1.s0 = 1u << 16;                                        // data_size=1 -> 2 bytes
  g1.s1 = (row_len_elems & 0xffffu) << 16;                 // tensor_dim0 [79:48] lo
  g1.s2 = (row_len_elems >> 16) | (64u << 16);             // tensor_dim0 hi | tensor_dim1 lo
  g1.s3 = (32u << 16);                                     // tensor_dim1 hi=0 | tile_dim0=32
  g1.s4 = 64u;                                             // tile_dim1=64, tile_dim2=0
  g1.s5 = (unsigned)(row_stride_elems & 0xffffffffu);      // tensor_dim0_stride lo
  g1.s6 = (unsigned)((row_stride_elems >> 32) & 0xffffu);  // stride hi | dim1_stride lo=0
  g1.s7 = 0u;
  asm volatile("tensor_load_to_lds %0, %1" :: "s"(g0), "s"(g1) : "memory");
}

// ---------------- f32 -> f16 convert ----------------
__global__ void cvt_f16_kernel(const float* __restrict__ in, _Float16* __restrict__ out, int n) {
  int i = blockIdx.x * 256 + threadIdx.x;
  if (i < n) out[i] = (_Float16)in[i];
}

// ---------------- weight convert + transpose: W[K][N] f32 -> Wt[N][K] f16 ----------------
__global__ void wt_kernel(const float* __restrict__ W, _Float16* __restrict__ Wt, int Kd, int Nd) {
  int idx = blockIdx.x * 256 + threadIdx.x;
  if (idx >= Kd * Nd) return;
  int k = idx / Nd, n = idx - k * Nd;
  Wt[(size_t)n * Kd + k] = (_Float16)W[idx];
}

// ---------------- LayerNorm over C=768, one row per block ----------------
__global__ __launch_bounds__(256) void ln_kernel(const float* __restrict__ X,
                                                 const float* __restrict__ g,
                                                 const float* __restrict__ bta,
                                                 _Float16* __restrict__ Y) {
  const int row = blockIdx.x, tid = threadIdx.x;
  __shared__ float red[256];
  const float* xr = X + (size_t)row * C_DIM;
  float v0 = xr[tid], v1 = xr[tid + 256], v2 = xr[tid + 512];
  red[tid] = v0 + v1 + v2;
  __syncthreads();
  for (int st = 128; st > 0; st >>= 1) { if (tid < st) red[tid] += red[tid + st]; __syncthreads(); }
  float mu = red[0] * (1.0f / C_DIM);
  __syncthreads();
  float d0 = v0 - mu, d1 = v1 - mu, d2 = v2 - mu;
  red[tid] = d0 * d0 + d1 * d1 + d2 * d2;
  __syncthreads();
  for (int st = 128; st > 0; st >>= 1) { if (tid < st) red[tid] += red[tid + st]; __syncthreads(); }
  float rstd = rsqrtf(red[0] * (1.0f / C_DIM) + 1e-5f);
  _Float16* yr = Y + (size_t)row * C_DIM;
  yr[tid]       = (_Float16)(d0 * rstd * g[tid]       + bta[tid]);
  yr[tid + 256] = (_Float16)(d1 * rstd * g[tid + 256] + bta[tid + 256]);
  yr[tid + 512] = (_Float16)(d2 * rstd * g[tid + 512] + bta[tid + 512]);
}

// ---------------- WMMA GEMM with TDM-staged, double-buffered LDS tiles ----------
// out = act(A[M,K] @ W[K,N] + bias), W given as Wt[N][K]
// MODE 0: f16 out; MODE 1: f16 gelu(out); MODE 2: f32 out = resid + acc + bias
template <int MODE>
__global__ __launch_bounds__(128) void gemm_kernel(const _Float16* __restrict__ A,
                                                   const _Float16* __restrict__ Wt,
                                                   const float* __restrict__ bias,
                                                   const float* __restrict__ resid,
                                                   void* __restrict__ outp,
                                                   int Md, int Nd, int Kd) {
  __shared__ __align__(32) _Float16 As[2][64][32];
  __shared__ __align__(32) _Float16 Bs[2][64][32];
  const int tid = threadIdx.x, lane = tid & 31, wave = tid >> 5;
  const int hi = lane >> 4, ln = lane & 15;
  const int m_base = blockIdx.y * 64, n_base = blockIdx.x * 64;
  const int msub = wave >> 1, nsub = wave & 1;
  const bool issuer = (wave == 0);
  float8 c[2][2] = {};

  const unsigned ldsA[2] = { (unsigned)(uintptr_t)(void*)&As[0][0][0],
                             (unsigned)(uintptr_t)(void*)&As[1][0][0] };
  const unsigned ldsB[2] = { (unsigned)(uintptr_t)(void*)&Bs[0][0][0],
                             (unsigned)(uintptr_t)(void*)&Bs[1][0][0] };
  const unsigned long long gA = (unsigned long long)(uintptr_t)(A  + (size_t)m_base * Kd);
  const unsigned long long gB = (unsigned long long)(uintptr_t)(Wt + (size_t)n_base * Kd);

  // prologue: DMA first K-tile into buffer 0
  if (issuer) {
    tdm_load_tile_64x32(ldsA[0], gA, (unsigned)Kd, (unsigned long long)Kd);
    tdm_load_tile_64x32(ldsB[0], gB, (unsigned)Kd, (unsigned long long)Kd);
  }

  int buf = 0;
  for (int k0 = 0; k0 < Kd; k0 += 32, buf ^= 1) {
    if (issuer) {
      if (k0 + 32 < Kd) {
        // overlap: DMA next tile into the alternate buffer, then wait only for
        // the current tile's two tensor ops (in-order completion per wave)
        tdm_load_tile_64x32(ldsA[buf ^ 1], gA + (size_t)(k0 + 32) * 2, (unsigned)Kd, (unsigned long long)Kd);
        tdm_load_tile_64x32(ldsB[buf ^ 1], gB + (size_t)(k0 + 32) * 2, (unsigned)Kd, (unsigned long long)Kd);
        __builtin_amdgcn_s_wait_tensorcnt(2);
      } else {
        __builtin_amdgcn_s_wait_tensorcnt(0);
      }
    }
    __syncthreads();

    half16 a[2], b[2];
#pragma unroll
    for (int mi = 0; mi < 2; mi++) {
      int r = msub * 32 + mi * 16 + ln;
      half8 alo = *(const half8*)&As[buf][r][hi * 8];
      half8 ahi = *(const half8*)&As[buf][r][hi * 8 + 16];
#pragma unroll
      for (int e = 0; e < 8; e++) { a[mi][e] = alo[e]; a[mi][8 + e] = ahi[e]; }
    }
#pragma unroll
    for (int ni = 0; ni < 2; ni++) {
      int r = nsub * 32 + ni * 16 + ln;
      half8 blo = *(const half8*)&Bs[buf][r][hi * 8];
      half8 bhi = *(const half8*)&Bs[buf][r][hi * 8 + 16];
#pragma unroll
      for (int e = 0; e < 8; e++) { b[ni][e] = blo[e]; b[ni][8 + e] = bhi[e]; }
    }
#pragma unroll
    for (int mi = 0; mi < 2; mi++)
#pragma unroll
      for (int ni = 0; ni < 2; ni++)
        c[mi][ni] = wmma16(a[mi], b[ni], c[mi][ni]);
    __syncthreads();  // everyone done reading buf before its next DMA overwrite
  }

#pragma unroll
  for (int mi = 0; mi < 2; mi++)
#pragma unroll
    for (int ni = 0; ni < 2; ni++) {
      int gm0 = m_base + msub * 32 + mi * 16 + hi * 8;
      int gn  = n_base + nsub * 32 + ni * 16 + ln;
      float bv = bias[gn];
#pragma unroll
      for (int i = 0; i < 8; i++) {
        size_t o = (size_t)(gm0 + i) * Nd + gn;
        float v = c[mi][ni][i] + bv;
        if (MODE == 0) {
          ((_Float16*)outp)[o] = (_Float16)v;
        } else if (MODE == 1) {
          ((_Float16*)outp)[o] = (_Float16)(0.5f * v * (1.0f + erff(v * 0.70710678f)));
        } else {
          ((float*)outp)[o] = resid[o] + v;
        }
      }
    }
}

// ---------------- Flash attention ----------------
// Block = 4 waves; each wave owns one 16-row Q tile. 64-key tiles are staged
// cooperatively once per block (K raw, V transposed) and shared through LDS;
// softmax stats amortized over 64 keys, reductions via v_permlane16.
__global__ __launch_bounds__(128) void flash_kernel(const _Float16* __restrict__ Q,
                                                    const _Float16* __restrict__ K,
                                                    const _Float16* __restrict__ V,
                                                    _Float16* __restrict__ Y,
                                                    int Tq, int Tk, float scale) {
  const int tid = threadIdx.x;
  const int lane = tid & 31, wave = tid >> 5;
  const int hi = lane >> 4, ln = lane & 15;
  const int b = blockIdx.z, h = blockIdx.y;
  const int q0 = blockIdx.x * 64 + wave * 16;

  __shared__ __align__(32) _Float16 Ks[64][64];       // [key][dh] raw K tile (8KB)
  __shared__ __align__(32) _Float16 VtS[64][64];      // [dh][key] transposed V tile (8KB)
  __shared__ __align__(32) _Float16 Pbuf[4][16][64];  // per-wave P tile 16x64 (8KB)

  const _Float16* qp = Q + ((size_t)(b * Tq + q0)) * C_DIM + h * DH;
  const _Float16* kp = K + ((size_t)b * Tk) * C_DIM + h * DH;
  const _Float16* vp = V + ((size_t)b * Tk) * C_DIM + h * DH;

  // Q as two A-fragments (dh 0..31, 32..63), resident for the whole K loop
  half16 aq[2];
#pragma unroll
  for (int s = 0; s < 2; s++) {
    const _Float16* base = qp + (size_t)ln * C_DIM + s * 32 + hi * 8;
    half8 lo = *(const half8*)base;
    half8 hv = *(const half8*)(base + 16);
#pragma unroll
    for (int e = 0; e < 8; e++) { aq[s][e] = lo[e]; aq[s][8 + e] = hv[e]; }
  }

  // staging roles (whole block, 128 threads)
  const int kRow = tid >> 1, kSeg = tid & 1;   // K copy: row 0..63, 32-elem segment 0..1
  const int vQuad = tid & 15, vDg = tid >> 4;  // V transpose: key quad 0..15, dh octet 0..7

  float m[8], l[8];
  float8 o[4] = {};
#pragma unroll
  for (int i = 0; i < 8; i++) { m[i] = -1e30f; l[i] = 0.0f; }

  for (int kt = 0; kt < Tk; kt += 64) {
    __syncthreads();  // previous tile's LDS reads complete before overwrite

    // ---- cooperative staging: K raw copy (4 x b128 per thread) ----
    {
      const _Float16* src = kp + (size_t)(kt + kRow) * C_DIM + kSeg * 32;
      _Float16* dst = &Ks[kRow][kSeg * 32];
#pragma unroll
      for (int c8 = 0; c8 < 4; c8++)
        *(half8*)(dst + c8 * 8) = *(const half8*)(src + c8 * 8);
    }
    // ---- cooperative staging: V transposed via 4x8 register blocks ----
    {
      half8 rv[4];
#pragma unroll
      for (int r = 0; r < 4; r++)
        rv[r] = *(const half8*)(vp + (size_t)(kt + vQuad * 4 + r) * C_DIM + vDg * 8);
#pragma unroll
      for (int e = 0; e < 8; e++) {
        half4 cv = { rv[0][e], rv[1][e], rv[2][e], rv[3][e] };
        *(half4*)&VtS[vDg * 8 + e][vQuad * 4] = cv;
      }
    }
    if (kt + 64 < Tk) {  // hint next key tile into cache
      __builtin_prefetch(kp + (size_t)(kt + 64 + kRow) * C_DIM);
      __builtin_prefetch(vp + (size_t)(kt + 64 + kRow) * C_DIM);
    }
    __syncthreads();

    // ---- S = Q K^T for 64 keys: four 16x16 tiles ----
    float8 sS[4] = {};
#pragma unroll
    for (int s = 0; s < 2; s++)
#pragma unroll
      for (int j = 0; j < 4; j++) {
        half8 klo = *(const half8*)&Ks[j * 16 + ln][s * 32 + hi * 8];
        half8 khi = *(const half8*)&Ks[j * 16 + ln][s * 32 + hi * 8 + 16];
        half16 bk;
#pragma unroll
        for (int e = 0; e < 8; e++) { bk[e] = klo[e]; bk[8 + e] = khi[e]; }
        sS[j] = wmma16(aq[s], bk, sS[j]);
      }

    // ---- online softmax across 64 columns ----
#pragma unroll
    for (int i = 0; i < 8; i++) {
      float sv[4];
#pragma unroll
      for (int j = 0; j < 4; j++) sv[j] = sS[j][i] * scale;
      float mx = fmaxf(fmaxf(sv[0], sv[1]), fmaxf(sv[2], sv[3]));
      XOR16(fmaxf, mx);
      float mn = fmaxf(m[i], mx);
      float alpha = __expf(m[i] - mn);
      m[i] = mn;
      float p[4], rs = 0.0f;
#pragma unroll
      for (int j = 0; j < 4; j++) { p[j] = __expf(sv[j] - mn); rs += p[j]; }
      XOR16(fadd_, rs);
      l[i] = l[i] * alpha + rs;
#pragma unroll
      for (int t = 0; t < 4; t++) o[t][i] *= alpha;
#pragma unroll
      for (int j = 0; j < 4; j++)
        Pbuf[wave][i + hi * 8][j * 16 + ln] = (_Float16)p[j];
    }

    // ---- O += P V : K=64 contraction as two 16x16x32 steps ----
#pragma unroll
    for (int s = 0; s < 2; s++) {
      half16 ap;
      {
        half8 plo = *(const half8*)&Pbuf[wave][ln][s * 32 + hi * 8];
        half8 phi = *(const half8*)&Pbuf[wave][ln][s * 32 + hi * 8 + 16];
#pragma unroll
        for (int e = 0; e < 8; e++) { ap[e] = plo[e]; ap[8 + e] = phi[e]; }
      }
#pragma unroll
      for (int t = 0; t < 4; t++) {
        half8 vlo = *(const half8*)&VtS[t * 16 + ln][s * 32 + hi * 8];
        half8 vhi = *(const half8*)&VtS[t * 16 + ln][s * 32 + hi * 8 + 16];
        half16 bv;
#pragma unroll
        for (int e = 0; e < 8; e++) { bv[e] = vlo[e]; bv[8 + e] = vhi[e]; }
        o[t] = wmma16(ap, bv, o[t]);
      }
    }
  }

  _Float16* yp = Y + ((size_t)(b * Tq + q0)) * C_DIM + h * DH;
#pragma unroll
  for (int i = 0; i < 8; i++) {
    float inv = 1.0f / l[i];
#pragma unroll
    for (int t = 0; t < 4; t++)
      yp[(size_t)(i + hi * 8) * C_DIM + t * 16 + ln] = (_Float16)(o[t][i] * inv);
  }
}

// ---------------- host side ----------------
extern "C" void kernel_launch(void* const* d_in, const int* in_sizes, int n_in,
                              void* d_out, int out_size, void* d_ws, size_t ws_size,
                              hipStream_t stream) {
  const float* x0   = (const float*)d_in[0];
  const float* ctx  = (const float*)d_in[1];
  const float* ln1g = (const float*)d_in[2]; const float* ln1b = (const float*)d_in[3];
  const float* ln2g = (const float*)d_in[4]; const float* ln2b = (const float*)d_in[5];
  const float* ln3g = (const float*)d_in[6]; const float* ln3b = (const float*)d_in[7];
  const float* W32[10] = { (const float*)d_in[8],  (const float*)d_in[10], (const float*)d_in[12],
                           (const float*)d_in[14], (const float*)d_in[16], (const float*)d_in[18],
                           (const float*)d_in[20], (const float*)d_in[22], (const float*)d_in[24],
                           (const float*)d_in[26] };
  const float* Bvec[10] = { (const float*)d_in[9],  (const float*)d_in[11], (const float*)d_in[13],
                            (const float*)d_in[15], (const float*)d_in[17], (const float*)d_in[19],
                            (const float*)d_in[21], (const float*)d_in[23], (const float*)d_in[25],
                            (const float*)d_in[27] };
  const int WK[10] = { C_DIM, C_DIM, C_DIM, C_DIM, C_DIM, C_DIM, C_DIM, C_DIM, C_DIM, F_DIM };
  const int WN[10] = { C_DIM, C_DIM, C_DIM, C_DIM, C_DIM, C_DIM, C_DIM, C_DIM, F_DIM, C_DIM };

  char* ws = (char*)d_ws;
  size_t off = 0;
  auto alloc = [&](size_t bytes) -> char* {
    off = (off + 255) & ~(size_t)255;
    char* p = ws + off;
    off += bytes;
    return p;
  };
  _Float16* Wt[10];
  for (int i = 0; i < 10; i++) Wt[i] = (_Float16*)alloc((size_t)WK[i] * WN[i] * 2);
  _Float16* ctx_h = (_Float16*)alloc((size_t)MROWS * C_DIM * 2);
  _Float16* xln   = (_Float16*)alloc((size_t)MROWS * C_DIM * 2);
  _Float16* qb    = (_Float16*)alloc((size_t)MROWS * C_DIM * 2);
  _Float16* kb    = (_Float16*)alloc((size_t)MROWS * C_DIM * 2);
  _Float16* vb    = (_Float16*)alloc((size_t)MROWS * C_DIM * 2);
  _Float16* yb    = (_Float16*)alloc((size_t)MROWS * C_DIM * 2);
  _Float16* hb    = (_Float16*)alloc((size_t)MROWS * F_DIM * 2);
  float*    x1b   = (float*)alloc((size_t)MROWS * C_DIM * 4);
  float*    x2b   = (float*)alloc((size_t)MROWS * C_DIM * 4);

  const dim3 blk128(128), blk256(256);
  const dim3 gC(C_DIM / 64, MROWS / 64);      // GEMM grid, N=768
  const dim3 gF(F_DIM / 64, MROWS / 64);      // GEMM grid, N=3072
  const dim3 gAtt(TSEQ / 64, NHEAD, BATCH);   // flash grid
  const float scale = 0.125f;                 // 1/sqrt(64)

  // one-time converts
  {
    int n = MROWS * C_DIM;
    cvt_f16_kernel<<<(n + 255) / 256, blk256, 0, stream>>>(ctx, ctx_h, n);
    for (int i = 0; i < 10; i++) {
      int nw = WK[i] * WN[i];
      wt_kernel<<<(nw + 255) / 256, blk256, 0, stream>>>(W32[i], Wt[i], WK[i], WN[i]);
    }
  }

  // ---- self-attention ----
  ln_kernel<<<MROWS, blk256, 0, stream>>>(x0, ln1g, ln1b, xln);
  gemm_kernel<0><<<gC, blk128, 0, stream>>>(xln, Wt[0], Bvec[0], nullptr, qb, MROWS, C_DIM, C_DIM);
  gemm_kernel<0><<<gC, blk128, 0, stream>>>(xln, Wt[1], Bvec[1], nullptr, kb, MROWS, C_DIM, C_DIM);
  gemm_kernel<0><<<gC, blk128, 0, stream>>>(xln, Wt[2], Bvec[2], nullptr, vb, MROWS, C_DIM, C_DIM);
  flash_kernel<<<gAtt, blk128, 0, stream>>>(qb, kb, vb, yb, TSEQ, TSEQ, scale);
  gemm_kernel<2><<<gC, blk128, 0, stream>>>(yb, Wt[3], Bvec[3], x0, x1b, MROWS, C_DIM, C_DIM);

  // ---- cross-attention (context not normalized, per reference) ----
  ln_kernel<<<MROWS, blk256, 0, stream>>>(x1b, ln2g, ln2b, xln);
  gemm_kernel<0><<<gC, blk128, 0, stream>>>(xln,   Wt[4], Bvec[4], nullptr, qb, MROWS, C_DIM, C_DIM);
  gemm_kernel<0><<<gC, blk128, 0, stream>>>(ctx_h, Wt[5], Bvec[5], nullptr, kb, MROWS, C_DIM, C_DIM);
  gemm_kernel<0><<<gC, blk128, 0, stream>>>(ctx_h, Wt[6], Bvec[6], nullptr, vb, MROWS, C_DIM, C_DIM);
  flash_kernel<<<gAtt, blk128, 0, stream>>>(qb, kb, vb, yb, TSEQ, TSEQ, scale);
  gemm_kernel<2><<<gC, blk128, 0, stream>>>(yb, Wt[7], Bvec[7], x1b, x2b, MROWS, C_DIM, C_DIM);

  // ---- MLP ----
  ln_kernel<<<MROWS, blk256, 0, stream>>>(x2b, ln3g, ln3b, xln);
  gemm_kernel<1><<<gF, blk128, 0, stream>>>(xln, Wt[8], Bvec[8], nullptr, hb, MROWS, F_DIM, C_DIM);
  gemm_kernel<2><<<gC, blk128, 0, stream>>>(hb, Wt[9], Bvec[9], x2b, (float*)d_out, MROWS, C_DIM, F_DIM);

  // ---- context passthrough (second tuple element) ----
  hipMemcpyAsync((float*)d_out + (size_t)MROWS * C_DIM, ctx,
                 (size_t)MROWS * C_DIM * sizeof(float), hipMemcpyDeviceToDevice, stream);
}
